// VGAE_all_13640816132538
// MI455X (gfx1250) — matrix operand order repaired and checked by reading, not compiled, not run on hardware
//
#include <hip/hip_runtime.h>
#include <hip/hip_bf16.h>

// ---------------------------------------------------------------------------
// VGAE forward for MI455X (gfx1250).
//
// Heavy op (NNConv edge-conditioned matmul) is restructured as 33 shared-B
// GEMMs per layer and mapped onto v_wmma_f32_16x16x32_bf16 (bf16 A/B, fp32
// accumulate). W = e@w2+b2 is never materialized (saves ~1.3 GB of traffic
// vs the reference dataflow). Everything else is fp32 VALU.
// nOC is a template parameter so accumulators are statically indexed
// (avoids v_movrels/v_movreld dynamic-VGPR indexing in the hot loop).
// ---------------------------------------------------------------------------

#define N_NODES   10000
#define N_EDGES   20000
#define IN_DIM    64
#define HID_DIM   64
#define LAT_DIM   32
#define EDGE_FEAT 16
#define HID_EDGE  32
#define N_GRAPHS  64
#define N_CLASSES 6
#define HID_CLS   64
#define KE        33   // HID_EDGE + 1 (bias folded in as k=32 with ek=1)

typedef __attribute__((ext_vector_type(16))) __bf16 v16bf;
typedef __attribute__((ext_vector_type(8)))  float  v8f;

// ------------------------- edge MLP: ek = relu(ea@w1+b1), ek[:,32]=1 -------
__global__ __launch_bounds__(64)
void edge_mlp_kernel(const float* __restrict__ ea,   // [E,16]
                     const float* __restrict__ w1,   // [16,32]
                     const float* __restrict__ b1,   // [32]
                     float* __restrict__ ek)         // [E,33]
{
    int e = blockIdx.x;
    int t = threadIdx.x;
    __shared__ float a[EDGE_FEAT];
    if (t < EDGE_FEAT) a[t] = ea[(size_t)e * EDGE_FEAT + t];
    __syncthreads();
    if (t < HID_EDGE) {
        float s = b1[t];
#pragma unroll
        for (int i = 0; i < EDGE_FEAT; ++i) s += a[i] * w1[i * HID_EDGE + t];
        ek[(size_t)e * KE + t] = fmaxf(s, 0.f);
    } else if (t == HID_EDGE) {
        ek[(size_t)e * KE + HID_EDGE] = 1.f;
    }
}

// --------- pack w2 (+b2 as k=32) into WMMA-B-fragment-ready bf16 layout ----
// w2p[(((k*2+c)*nOC+oc)*32+lane)*16 + j] = ww[k][i = c*32 + 16*half + j][oc*16+n]
// with lane = (half,n); contiguous k order K = 16*half + j, N = n (same order as the A fragments).
__global__ void pack_w2_kernel(const float* __restrict__ w2,  // [32, 64*out]
                               const float* __restrict__ b2,  // [64*out]
                               __hip_bfloat16* __restrict__ w2p,
                               int out_dim)
{
    int nOC  = out_dim >> 4;
    int idx  = blockIdx.x * blockDim.x + threadIdx.x;
    int total = KE * 2 * nOC * 32;
    if (idx >= total) return;
    int lane = idx & 31;
    int tmp  = idx >> 5;
    int oc   = tmp % nOC; tmp /= nOC;
    int c    = tmp & 1;
    int k    = tmp >> 1;
    int half = lane >> 4;
    int n    = lane & 15;
    int o    = oc * 16 + n;
    __hip_bfloat16* d = w2p + (size_t)idx * 16;
#pragma unroll
    for (int j = 0; j < 16; ++j) {
        int i = c * 32 + 16 * half + j;  // in-dim (K) index
        float v = (k < HID_EDGE)
                    ? w2[(size_t)k * IN_DIM * out_dim + (size_t)i * out_dim + o]
                    : b2[(size_t)i * out_dim + o];
        d[j] = __float2bfloat16(v);
    }
}

// --------------------------- in-degree counts ------------------------------
__global__ void deg_kernel(const int* __restrict__ dst, float* __restrict__ deg, int E)
{
    int e = blockIdx.x * blockDim.x + threadIdx.x;
    if (e < E) atomicAdd(&deg[dst[e]], 1.f);
}

// --------------------- fused edge-message kernel (WMMA core) ---------------
// One wave handles a tile of 16 edges:
//   msg[e,o] = sum_k ek[e,k] * (h[src[e],:] @ ww[k])[o]
// then scatter-adds msg into agg[dst[e], o] with fp32 atomics.
// NOC = out_dim/16 (compile-time so acc[] is statically register-indexed).
template <int NOC>
__global__ __launch_bounds__(32)
void nnconv_msg_kernel(const float* __restrict__ hin,            // [N,64]
                       const float* __restrict__ ek,             // [E,33]
                       const __hip_bfloat16* __restrict__ w2p,   // packed B frags
                       const int* __restrict__ src,
                       const int* __restrict__ dst,
                       float* __restrict__ agg)                  // [N, NOC*16]
{
    constexpr int OUT = NOC * 16;
    const int e0   = blockIdx.x * 16;
    const int lane = threadIdx.x;
    const int half = lane >> 4;
    const int nrow = lane & 15;

    __shared__ float hs[16][IN_DIM];   // gathered h[src] rows
    __shared__ float eks[16][KE];
    __shared__ int   dsts[16];

    // stage h[src] tile: lane -> (edge t = lane/2, 32-col half)
    {
        int t  = lane >> 1;
        int cb = (lane & 1) * 32;
        int s  = src[e0 + t];
        const float4* gp = (const float4*)(hin + (size_t)s * IN_DIM + cb);
        float4* lp = (float4*)(&hs[t][cb]);
#pragma unroll
        for (int q = 0; q < 8; ++q) lp[q] = gp[q];
    }
    if (lane < 16) {
        dsts[lane] = dst[e0 + lane];
        const float* ep = ek + (size_t)(e0 + lane) * KE;
#pragma unroll
        for (int k = 0; k < KE; ++k) eks[lane][k] = ep[k];
    }
    __syncthreads();

    // Build A fragments (16x32 bf16, K = 16*half + j, matching packed B)
    v16bf A[2];
#pragma unroll
    for (int c = 0; c < 2; ++c)
#pragma unroll
        for (int j = 0; j < 16; ++j) {
            int K = 16 * half + j;
            A[c][j] = (__bf16)hs[nrow][c * 32 + K];
        }

    v8f acc[NOC] = {};   // msg accumulators, one per 16-col out chunk

    for (int k = 0; k < KE; ++k) {
        float ekv[8];
#pragma unroll
        for (int r = 0; r < 8; ++r) ekv[r] = eks[r + 8 * half][k];
#pragma unroll
        for (int oc = 0; oc < NOC; ++oc) {
            v8f q = {};
#pragma unroll
            for (int c = 0; c < 2; ++c) {
                const v16bf B = *(const v16bf*)(
                    w2p + ((((size_t)k * 2 + c) * NOC + oc) * 32 + lane) * 16);
                q = __builtin_amdgcn_wmma_f32_16x16x32_bf16(
                        false, A[c], false, B, (short)0, q, false, false);
            }
#pragma unroll
            for (int r = 0; r < 8; ++r) acc[oc][r] += ekv[r] * q[r];
        }
    }

    // scatter-reduce: C/D layout row M = r + 8*half, col N = nrow
#pragma unroll
    for (int oc = 0; oc < NOC; ++oc) {
        int o = oc * 16 + nrow;
#pragma unroll
        for (int r = 0; r < 8; ++r) {
            int node = dsts[r + 8 * half];
            atomicAdd(&agg[(size_t)node * OUT + o], acc[oc][r]);
        }
    }
}

// ------------- node update: hout = act(hin@root + bias + agg[/deg]) --------
template <int OUT>
__global__ void node_update_kernel(const float* __restrict__ hin,   // [N,64]
                                   const float* __restrict__ root,  // [64,OUT]
                                   const float* __restrict__ bias,  // [OUT]
                                   const float* __restrict__ agg,   // [N,OUT]
                                   const float* __restrict__ deg,   // [N] or null
                                   float* __restrict__ hout,        // [N,OUT]
                                   int do_relu)
{
    int n = blockIdx.x;
    int o = threadIdx.x;                 // blockDim == OUT (32 or 64)
    __shared__ float hrow[IN_DIM];
#pragma unroll
    for (int i = o; i < IN_DIM; i += OUT) hrow[i] = hin[(size_t)n * IN_DIM + i];
    __syncthreads();
    float s = bias[o];
#pragma unroll
    for (int i = 0; i < IN_DIM; ++i) s += hrow[i] * root[i * OUT + o];
    float a = agg[(size_t)n * OUT + o];
    if (deg) a /= fmaxf(deg[n], 1.f);
    s += a;
    hout[(size_t)n * OUT + o] = do_relu ? fmaxf(s, 0.f) : s;
}

// ----------------------------- reparametrize -------------------------------
__global__ void reparam_kernel(const float* __restrict__ mu,
                               const float* __restrict__ logvar,
                               const float* __restrict__ eps,
                               float* __restrict__ z, int n)
{
    int i = blockIdx.x * blockDim.x + threadIdx.x;
    if (i < n) {
        float lv = fminf(fmaxf(logvar[i], -5.f), 5.f);
        z[i] = mu[i] + eps[i] * expf(0.5f * lv);
    }
}

// ----------------------- edge decoder: sigmoid(z_s . z_d) ------------------
__global__ void adj_kernel(const float* __restrict__ z,
                           const int* __restrict__ src, const int* __restrict__ dst,
                           float* __restrict__ adj, int E)
{
    int e = blockIdx.x * blockDim.x + threadIdx.x;
    if (e < E) {
        const float* a = z + (size_t)src[e] * LAT_DIM;
        const float* b = z + (size_t)dst[e] * LAT_DIM;
        float s = 0.f;
#pragma unroll
        for (int d = 0; d < LAT_DIM; ++d) s += a[d] * b[d];
        adj[e] = 1.f / (1.f + expf(-s));
    }
}

// ----------------------------- global mean pool ----------------------------
__global__ void pool_kernel(const float* __restrict__ z, const int* __restrict__ batch,
                            float* __restrict__ gsum, float* __restrict__ gcnt, int n)
{
    int i = blockIdx.x * blockDim.x + threadIdx.x;
    if (i < n * LAT_DIM) {
        int node = i >> 5, d = i & (LAT_DIM - 1);
        int g = batch[node];
        atomicAdd(&gsum[g * LAT_DIM + d], z[i]);
        if (d == 0) atomicAdd(&gcnt[g], 1.f);
    }
}

// ------------------------------- classifier --------------------------------
__global__ __launch_bounds__(64)
void cls_kernel(const float* __restrict__ gsum, const float* __restrict__ gcnt,
                const float* __restrict__ w1, const float* __restrict__ b1,  // [32,64],[64]
                const float* __restrict__ w2, const float* __restrict__ b2,  // [64,6],[6]
                float* __restrict__ logits)                                  // [G,6]
{
    int g = blockIdx.x;
    int t = threadIdx.x;
    __shared__ float ge[LAT_DIM];
    __shared__ float hc[HID_CLS];
    float cnt = fmaxf(gcnt[g], 1.f);
    if (t < LAT_DIM) ge[t] = gsum[g * LAT_DIM + t] / cnt;
    __syncthreads();
    float s = b1[t];
#pragma unroll
    for (int d = 0; d < LAT_DIM; ++d) s += ge[d] * w1[d * HID_CLS + t];
    hc[t] = fmaxf(s, 0.f);
    __syncthreads();
    if (t < N_CLASSES) {
        float s2 = b2[t];
#pragma unroll
        for (int d = 0; d < HID_CLS; ++d) s2 += hc[d] * w2[d * N_CLASSES + t];
        logits[g * N_CLASSES + t] = s2;
    }
}

// ===========================================================================
extern "C" void kernel_launch(void* const* d_in, const int* in_sizes, int n_in,
                              void* d_out, int out_size, void* d_ws, size_t ws_size,
                              hipStream_t stream)
{
    // -------- inputs (setup_inputs() dict order; params flattened in dict
    //          insertion order: w1,b1,w2,b2,root,bias per conv) --------------
    const float* x         = (const float*)d_in[0];
    const float* edge_attr = (const float*)d_in[1];
    const float* eps       = (const float*)d_in[2];
    const float* cp[4][6];                 // conv1, conv2, conv_mu, conv_logvar
    for (int ci = 0; ci < 4; ++ci)
        for (int pi = 0; pi < 6; ++pi)
            cp[ci][pi] = (const float*)d_in[3 + ci * 6 + pi];
    const float* cls_w1 = (const float*)d_in[27];
    const float* cls_b1 = (const float*)d_in[28];
    const float* cls_w2 = (const float*)d_in[29];
    const float* cls_b2 = (const float*)d_in[30];
    const int*   edge_index = (const int*)d_in[31];
    const int*   batch      = (const int*)d_in[32];
    const int*   src = edge_index;              // row 0
    const int*   dst = edge_index + N_EDGES;    // row 1

    // -------- outputs (return-order concat) ---------------------------------
    float* out     = (float*)d_out;
    float* adj     = out;                        // 20000
    float* mu      = adj + N_EDGES;              // 320000
    float* logvar  = mu + N_NODES * LAT_DIM;     // 320000
    float* logits  = logvar + N_NODES * LAT_DIM; // 384
    float* z       = logits + N_GRAPHS * N_CLASSES; // 320000

    // -------- workspace carve (256B aligned) --------------------------------
    char* w = (char*)d_ws;
    size_t off = 0;
    auto carve = [&](size_t bytes) {
        void* p = w + off;
        off = (off + bytes + 255) & ~(size_t)255;
        return p;
    };
    float*          ek   = (float*)carve((size_t)N_EDGES * KE * 4);        // 2.64 MB
    __hip_bfloat16* w2p  = (__hip_bfloat16*)carve((size_t)KE * 2 * 4 * 32 * 16 * 2); // 270 KB
    float*          h1   = (float*)carve((size_t)N_NODES * HID_DIM * 4);   // 2.56 MB
    float*          h2   = (float*)carve((size_t)N_NODES * HID_DIM * 4);
    float*          agg  = (float*)carve((size_t)N_NODES * HID_DIM * 4);
    float*          deg  = (float*)carve((size_t)N_NODES * 4);
    float*          gsum = (float*)carve((size_t)N_GRAPHS * LAT_DIM * 4);
    float*          gcnt = (float*)carve((size_t)N_GRAPHS * 4);
    (void)ws_size; (void)in_sizes; (void)n_in; (void)out_size;

    // -------- in-degree (shared by the two 'mean' layers) -------------------
    (void)hipMemsetAsync(deg, 0, (size_t)N_NODES * 4, stream);
    deg_kernel<<<(N_EDGES + 255) / 256, 256, 0, stream>>>(dst, deg, N_EDGES);

    // -------- one NNConv layer (compile-time out width) ---------------------
    auto run_conv = [&]<int NOC>(const float* hin, const float* const* p,
                                 bool mean, bool relu, float* hout) {
        constexpr int OUT = NOC * 16;
        edge_mlp_kernel<<<N_EDGES, 64, 0, stream>>>(edge_attr, p[0], p[1], ek);
        constexpr int total = KE * 2 * NOC * 32;
        pack_w2_kernel<<<(total + 63) / 64, 64, 0, stream>>>(p[2], p[3], w2p, OUT);
        (void)hipMemsetAsync(agg, 0, (size_t)N_NODES * OUT * 4, stream);
        nnconv_msg_kernel<NOC><<<N_EDGES / 16, 32, 0, stream>>>(hin, ek, w2p,
                                                                src, dst, agg);
        node_update_kernel<OUT><<<N_NODES, OUT, 0, stream>>>(
            hin, p[4], p[5], agg, mean ? deg : nullptr, hout, relu ? 1 : 0);
    };

    run_conv.template operator()<4>(x,  cp[0], /*mean=*/true,  /*relu=*/true,  h1);
    run_conv.template operator()<4>(h1, cp[1], /*mean=*/true,  /*relu=*/true,  h2);
    run_conv.template operator()<2>(h2, cp[2], /*mean=*/false, /*relu=*/false, mu);
    run_conv.template operator()<2>(h2, cp[3], /*mean=*/false, /*relu=*/false, logvar);

    // -------- reparametrize, decode edges, pool, classify -------------------
    reparam_kernel<<<(N_NODES * LAT_DIM + 255) / 256, 256, 0, stream>>>(
        mu, logvar, eps, z, N_NODES * LAT_DIM);
    adj_kernel<<<(N_EDGES + 255) / 256, 256, 0, stream>>>(z, src, dst, adj, N_EDGES);
    (void)hipMemsetAsync(gsum, 0, (size_t)N_GRAPHS * LAT_DIM * 4, stream);
    (void)hipMemsetAsync(gcnt, 0, (size_t)N_GRAPHS * 4, stream);
    pool_kernel<<<(N_NODES * LAT_DIM + 255) / 256, 256, 0, stream>>>(
        z, batch, gsum, gcnt, N_NODES);
    cls_kernel<<<N_GRAPHS, 64, 0, stream>>>(gsum, gcnt, cls_w1, cls_b1,
                                            cls_w2, cls_b2, logits);
}
